// Decoder_84713934946433
// MI455X (gfx1250) — matrix operand order repaired and checked by reading, not compiled, not run on hardware
//
#include <hip/hip_runtime.h>
#include <cstdint>
#include <cstddef>

// MI455X / gfx1250: wave32, WMMA 16x16x32 bf16, 320KB LDS per WGP.
// Strategy: fuse entire 12-step decoder recurrence into one kernel.
// Encoder tile lives in LDS as bf16 (read from HBM exactly once),
// GRU h@W_hh.T runs on v_wmma_f32_16x16x32_bf16, everything else fp32 VALU.

#define T_IN   12
#define T_OUT  12
#define BATCH  65536
#define HDIM   64
#define ROWS   64          // batch rows per block
#define NTHR   64          // 2 waves of 32

typedef __attribute__((ext_vector_type(16))) __bf16 bf16x16;
typedef __attribute__((ext_vector_type(8)))  float  f32x8;

// ---- LDS byte map (dynamic shared) ----
#define ENC_OFF   0                          // bf16 [64][12*64]   98304 B
#define HID_OFF   (ENC_OFF  + ROWS*T_IN*HDIM*2)   // f32 [64][64]  16384 B
#define AST_OFF   (HID_OFF  + ROWS*HDIM*4)        // bf16 [64][64]  8192 B
#define WHH_OFF   (AST_OFF  + ROWS*HDIM*2)        // bf16 [192][64] 24576 B
#define WIH_OFF   (WHH_OFF  + 192*HDIM*2)         // f32 [192][4]   3072 B
#define BHH_OFF   (WIH_OFF  + 192*16)             // f32 [192]       768 B
#define LST_OFF   (BHH_OFF  + 192*4)              // f32 [64][4]    1024 B
#define OUTB_OFF  (LST_OFF  + ROWS*16)            // f32 [64]        256 B
#define WLIN_OFF  (OUTB_OFF + ROWS*4)             // f32 [64]        256 B
#define SMEM_BYTES (WLIN_OFF + HDIM*4)            // = 152832 B

__device__ __forceinline__ unsigned short f2bf(float f) {
  union { float f; unsigned u; } v; v.f = f;
  unsigned r = v.u + 0x7FFFu + ((v.u >> 16) & 1u);   // RNE
  return (unsigned short)(r >> 16);
}
__device__ __forceinline__ float bflo(unsigned u){ union{unsigned v;float f;}x; x.v = u << 16; return x.f; }
__device__ __forceinline__ float bfhi(unsigned u){ union{unsigned v;float f;}x; x.v = u & 0xffff0000u; return x.f; }
__device__ __forceinline__ float fsig(float x){ return 1.f/(1.f + __expf(-x)); }
__device__ __forceinline__ float ftanh(float x){ float e = __expf(2.f*x); return 1.f - 2.f/(e + 1.f); }

union ABfrag { uint4 u[2]; bf16x16 v; };

// gh tile for gate columns [col, col+16): D = A(16x64 bf16) x W_hhT(64x16 bf16), K split in two.
// B layout per ISA: lanes 0-15 hold K=0..7,16..23 (2 per VGPR), lanes 16-31 hold K=8..15,24..31.
__device__ __forceinline__ f32x8 gh_tile(const char* smem, const ABfrag& a0, const ABfrag& a1,
                                         int col, int hi, int n) {
  ABfrag b0, b1;
  const int base = WHH_OFF + (col + n)*128 + hi*16;   // [j][k] bf16, k contiguous
  b0.u[0] = *(const uint4*)&smem[base];        // K 0..31 step: octets hi, hi+2
  b0.u[1] = *(const uint4*)&smem[base + 32];
  b1.u[0] = *(const uint4*)&smem[base + 64];   // K 32..63 step (+32 K = +64 B)
  b1.u[1] = *(const uint4*)&smem[base + 96];
  f32x8 acc = {0.f,0.f,0.f,0.f,0.f,0.f,0.f,0.f};
  acc = __builtin_amdgcn_wmma_f32_16x16x32_bf16(false, a0.v, false, b0.v, (short)0, acc, false, false);
  acc = __builtin_amdgcn_wmma_f32_16x16x32_bf16(false, a1.v, false, b1.v, (short)0, acc, false, false);
  return acc;
}

__global__ __launch_bounds__(NTHR)
void decoder_gfx1250_kernel(const float* __restrict__ enc_g,
                            const float* __restrict__ ehid_g,
                            const float* __restrict__ last_g,
                            const float* __restrict__ wih_g,
                            const float* __restrict__ whh_g,
                            const float* __restrict__ bih_g,
                            const float* __restrict__ bhh_g,
                            const float* __restrict__ wlin_g,
                            const float* __restrict__ blin_g,
                            float* __restrict__ out_g)
{
  extern __shared__ char smem[];
  const int tid = threadIdx.x;
  const int bb  = blockIdx.x * ROWS;

  // ---- stage inputs into LDS (encoder_out read from HBM exactly once) ----
  for (int idx = tid; idx < ROWS*T_IN*HDIM; idx += NTHR) {
    int r = idx / (T_IN*HDIM); int rem = idx - r*(T_IN*HDIM);
    int t = rem >> 6; int k = rem & 63;
    float v = enc_g[((size_t)t*BATCH + bb + r)*HDIM + k];
    *(unsigned short*)&smem[ENC_OFF + (size_t)idx*2] = f2bf(v);
  }
  for (int idx = tid; idx < ROWS*HDIM; idx += NTHR)
    ((float*)&smem[HID_OFF])[idx] = ehid_g[(size_t)bb*HDIM + idx];
  for (int idx = tid; idx < 192*HDIM; idx += NTHR)
    *(unsigned short*)&smem[WHH_OFF + idx*2] = f2bf(whh_g[idx]);
  for (int j = tid; j < 192; j += NTHR) {
    float4 w; w.x = wih_g[j*3]; w.y = wih_g[j*3+1]; w.z = wih_g[j*3+2]; w.w = bih_g[j];
    *(float4*)&smem[WIH_OFF + j*16] = w;
    ((float*)&smem[BHH_OFF])[j] = bhh_g[j];
  }
  if (tid < HDIM) ((float*)&smem[WLIN_OFF])[tid] = wlin_g[tid];
  if (tid < ROWS) {
    float4 l; l.x = last_g[(size_t)(bb+tid)*3]; l.y = last_g[(size_t)(bb+tid)*3+1];
    l.z = last_g[(size_t)(bb+tid)*3+2]; l.w = 0.f;
    *(float4*)&smem[LST_OFF + tid*16] = l;
  }
  const float blin = blin_g[0];
  __syncthreads();

  const int lane   = tid & 31;
  const int waveId = tid >> 5;
  const int hi     = lane >> 4;   // lane>=16
  const int n16    = lane & 15;

  for (int step = 0; step < T_OUT; ++step) {
    // ================= Phase A: attention, one row per lane =================
    {
      const int row = tid;
      if (step > 0) {   // lst = [out_prev, lst0, lst1]
        float4 l = *(float4*)&smem[LST_OFF + row*16];
        float o  = ((float*)&smem[OUTB_OFF])[row];
        float4 nl; nl.x = o; nl.y = l.x; nl.z = l.y; nl.w = 0.f;
        *(float4*)&smem[LST_OFF + row*16] = nl;
      }
      float h[HDIM];
      #pragma unroll
      for (int c = 0; c < 16; ++c) {
        float4 v = *(float4*)&smem[HID_OFF + row*256 + c*16];
        h[c*4+0]=v.x; h[c*4+1]=v.y; h[c*4+2]=v.z; h[c*4+3]=v.w;
      }
      float s[T_IN];
      #pragma unroll
      for (int t = 0; t < T_IN; ++t) {
        float acc = 0.f;
        #pragma unroll
        for (int c = 0; c < 8; ++c) {
          uint4 e = *(uint4*)&smem[ENC_OFF + row*1536 + t*128 + c*16];
          acc = fmaf(bflo(e.x), h[c*8+0], acc); acc = fmaf(bfhi(e.x), h[c*8+1], acc);
          acc = fmaf(bflo(e.y), h[c*8+2], acc); acc = fmaf(bfhi(e.y), h[c*8+3], acc);
          acc = fmaf(bflo(e.z), h[c*8+4], acc); acc = fmaf(bfhi(e.z), h[c*8+5], acc);
          acc = fmaf(bflo(e.w), h[c*8+6], acc); acc = fmaf(bfhi(e.w), h[c*8+7], acc);
        }
        s[t] = acc;
      }
      float mx = s[0];
      #pragma unroll
      for (int t = 1; t < T_IN; ++t) mx = fmaxf(mx, s[t]);
      float den = 0.f;
      #pragma unroll
      for (int t = 0; t < T_IN; ++t) { s[t] = __expf(s[t]-mx); den += s[t]; }
      const float inv = 1.f/den;
      #pragma unroll
      for (int t = 0; t < T_IN; ++t) {
        float w = s[t]*inv;
        #pragma unroll
        for (int c = 0; c < 8; ++c) {
          uint4 e = *(uint4*)&smem[ENC_OFF + row*1536 + t*128 + c*16];
          h[c*8+0] = fmaf(w, bflo(e.x), h[c*8+0]); h[c*8+1] = fmaf(w, bfhi(e.x), h[c*8+1]);
          h[c*8+2] = fmaf(w, bflo(e.y), h[c*8+2]); h[c*8+3] = fmaf(w, bfhi(e.y), h[c*8+3]);
          h[c*8+4] = fmaf(w, bflo(e.z), h[c*8+4]); h[c*8+5] = fmaf(w, bfhi(e.z), h[c*8+5]);
          h[c*8+6] = fmaf(w, bflo(e.w), h[c*8+6]); h[c*8+7] = fmaf(w, bfhi(e.w), h[c*8+7]);
        }
      }
      #pragma unroll
      for (int c = 0; c < 16; ++c) {   // canonical f32 hidden
        float4 v; v.x=h[c*4]; v.y=h[c*4+1]; v.z=h[c*4+2]; v.w=h[c*4+3];
        *(float4*)&smem[HID_OFF + row*256 + c*16] = v;
      }
      #pragma unroll
      for (int c = 0; c < 8; ++c) {    // bf16 A-staging [row][k]
        uint4 u;
        u.x = (unsigned)f2bf(h[c*8+0]) | ((unsigned)f2bf(h[c*8+1])<<16);
        u.y = (unsigned)f2bf(h[c*8+2]) | ((unsigned)f2bf(h[c*8+3])<<16);
        u.z = (unsigned)f2bf(h[c*8+4]) | ((unsigned)f2bf(h[c*8+5])<<16);
        u.w = (unsigned)f2bf(h[c*8+6]) | ((unsigned)f2bf(h[c*8+7])<<16);
        *(uint4*)&smem[AST_OFF + row*128 + c*16] = u;
      }
    }
    __syncthreads();

    // ================= Phase B: GRU via WMMA (per wave: 32 rows) =================
    #pragma unroll 1
    for (int mt = 0; mt < 2; ++mt) {
      const int rowBase = waveId*32 + mt*16;
      const int m = rowBase + n16;           // A-frag M row for this lane
      // A layout per ISA: v0-3 = K group (hi?8:0)+0..7, v4-7 = +16 (hi alternates K octets)
      ABfrag a0, a1;
      a0.u[0] = *(uint4*)&smem[AST_OFF + m*128 +       hi*16];
      a0.u[1] = *(uint4*)&smem[AST_OFF + m*128 + 32 +  hi*16];
      a1.u[0] = *(uint4*)&smem[AST_OFF + m*128 + 64 +  hi*16];
      a1.u[1] = *(uint4*)&smem[AST_OFF + m*128 + 96 +  hi*16];
      float4 l4[8];
      #pragma unroll
      for (int r = 0; r < 8; ++r)
        l4[r] = *(float4*)&smem[LST_OFF + (rowBase + hi*8 + r)*16];
      float po[8];
      #pragma unroll
      for (int r = 0; r < 8; ++r) po[r] = 0.f;

      #pragma unroll 1
      for (int J = 0; J < 4; ++J) {
        const int colR = J*16, colZ = 64 + J*16, colN = 128 + J*16;
        f32x8 ghR = gh_tile(smem, a0, a1, colR, hi, n16);
        f32x8 ghZ = gh_tile(smem, a0, a1, colZ, hi, n16);
        f32x8 ghN = gh_tile(smem, a0, a1, colN, hi, n16);
        const float4 w4R = *(float4*)&smem[WIH_OFF + (colR+n16)*16];
        const float4 w4Z = *(float4*)&smem[WIH_OFF + (colZ+n16)*16];
        const float4 w4N = *(float4*)&smem[WIH_OFF + (colN+n16)*16];
        const float bhR = ((float*)&smem[BHH_OFF])[colR+n16];
        const float bhZ = ((float*)&smem[BHH_OFF])[colZ+n16];
        const float bhN = ((float*)&smem[BHH_OFF])[colN+n16];
        const float wl  = ((float*)&smem[WLIN_OFF])[J*16+n16];
        #pragma unroll
        for (int r = 0; r < 8; ++r) {   // D layout: vgpr r -> M = r + hi*8
          const int rowr = rowBase + hi*8 + r;
          float giR = fmaf(w4R.x,l4[r].x, fmaf(w4R.y,l4[r].y, fmaf(w4R.z,l4[r].z, w4R.w)));
          float giZ = fmaf(w4Z.x,l4[r].x, fmaf(w4Z.y,l4[r].y, fmaf(w4Z.z,l4[r].z, w4Z.w)));
          float giN = fmaf(w4N.x,l4[r].x, fmaf(w4N.y,l4[r].y, fmaf(w4N.z,l4[r].z, w4N.w)));
          float rr = fsig(ghR[r] + bhR + giR);
          float zz = fsig(ghZ[r] + bhZ + giZ);
          float nv = ftanh(fmaf(rr, ghN[r] + bhN, giN));   // n = tanh(i_n + r*(h_n + b_hh_n))
          float* hp = (float*)&smem[HID_OFF + rowr*256 + (J*16+n16)*4];
          float hold = *hp;
          float hnew = fmaf(zz, hold - nv, nv);            // (1-z)*n + z*h
          *hp = hnew;
          po[r] = fmaf(hnew, wl, po[r]);
        }
      }
      // reduce out-partials across the 16 N-lanes of each half
      #pragma unroll
      for (int r = 0; r < 8; ++r) {
        float v = po[r];
        v += __shfl_xor(v, 1, 32);
        v += __shfl_xor(v, 2, 32);
        v += __shfl_xor(v, 4, 32);
        v += __shfl_xor(v, 8, 32);
        po[r] = v;
      }
      if (n16 == 0) {
        #pragma unroll
        for (int r = 0; r < 8; ++r) {
          const int rowr = rowBase + hi*8 + r;
          const float val = po[r] + blin;
          ((float*)&smem[OUTB_OFF])[rowr] = val;                 // feeds next lst
          out_g[(size_t)(bb + rowr)*T_OUT + step] = val;         // (B, T_OUT) row-major
        }
      }
    }
    __syncthreads();
  }
}

extern "C" void kernel_launch(void* const* d_in, const int* in_sizes, int n_in,
                              void* d_out, int out_size, void* d_ws, size_t ws_size,
                              hipStream_t stream) {
  (void)in_sizes; (void)n_in; (void)out_size; (void)d_ws; (void)ws_size;
  const float* enc  = (const float*)d_in[0];
  const float* ehid = (const float*)d_in[1];
  const float* last = (const float*)d_in[2];
  const float* wih  = (const float*)d_in[3];
  const float* whh  = (const float*)d_in[4];
  const float* bih  = (const float*)d_in[5];
  const float* bhh  = (const float*)d_in[6];
  const float* wlin = (const float*)d_in[7];
  const float* blin = (const float*)d_in[8];
  float* out = (float*)d_out;

  // Deterministic, capture-safe: not a stream op, idempotent.
  hipFuncSetAttribute(reinterpret_cast<const void*>(decoder_gfx1250_kernel),
                      hipFuncAttributeMaxDynamicSharedMemorySize, SMEM_BYTES);

  dim3 grid(BATCH / ROWS);   // 1024 blocks of 64 threads (2 waves)
  decoder_gfx1250_kernel<<<grid, NTHR, SMEM_BYTES, stream>>>(
      enc, ehid, last, wih, whh, bih, bhh, wlin, blin, out);
}
